// LSTMAttentionCell_68624987455858
// MI455X (gfx1250) — matrix operand order repaired and checked by reading, not compiled, not run on hardware
//
#include <hip/hip_runtime.h>
#include <hip/hip_bf16.h>
#include <math.h>

// ---------------- problem constants ----------------
#define BATCH   128
#define TSTEPS  300
#define SLEN    64
#define NCHARS  57
#define KGAUSS  10
#define ULSTM   457            // NUM_CELLS + NUM_CHARS
#define N4      1828           // 4 * ULSTM
#define INDIM   60             // 3 strokes + 57 window chars
#define KDIM    517            // INDIM + ULSTM
#define KPAD    544            // KDIM padded to multiple of 32
#define KC      17             // KPAD / 32 k-chunks
#define NT      115            // ceil(N4 / 16) N-tiles
#define MT      8              // BATCH / 16 M-tiles

typedef __attribute__((ext_vector_type(16))) __bf16 v16bf;
typedef __attribute__((ext_vector_type(8)))  float  v8f;

// -------------------------------------------------------------------------
// Pack merged [Wx ; Wh] (fp32, row-major [K, N4]) into bf16 WMMA B-fragments.
// Fragment block (nt, kc): 32 lanes x 16 bf16, contiguous.
//   lane l: column n = nt*16 + (l & 15)
//   element e: k = kc*32 + e + 16*(l >= 16)
// Zero-pad k >= 517 and n >= 1828.
// -------------------------------------------------------------------------
__global__ __launch_bounds__(256) void pack_weights(
    const float* __restrict__ Wx, const float* __restrict__ Wh,
    __bf16* __restrict__ wpack) {
  int gid = blockIdx.x * 256 + threadIdx.x;
  if (gid >= NT * KC * 32) return;
  int lane = gid & 31;
  int unit = gid >> 5;
  int nt = unit / KC, kc = unit % KC;
  int n = nt * 16 + (lane & 15);
  int kbase = kc * 32 + ((lane >= 16) ? 16 : 0);
  __bf16* dst = wpack + (size_t)unit * 512 + lane * 16;
#pragma unroll
  for (int e = 0; e < 16; ++e) {
    int k = kbase + e;
    float v = 0.0f;
    if (n < N4) {
      if (k < INDIM)      v = Wx[(size_t)k * N4 + n];
      else if (k < KDIM)  v = Wh[(size_t)(k - INDIM) * N4 + n];
    }
    dst[e] = (__bf16)v;
  }
}

// -------------------------------------------------------------------------
// Initialize recurrent state: c=0, kappa=0, act row = [x_0, w=0, h=0, pad=0]
// act: [BATCH][KPAD] fp32
// -------------------------------------------------------------------------
__global__ __launch_bounds__(256) void init_state(
    const float* __restrict__ inputs, float* __restrict__ act,
    float* __restrict__ cstate, float* __restrict__ kstate) {
  int gid = blockIdx.x * 256 + threadIdx.x;
  if (gid < BATCH * KPAD) {
    int b = gid / KPAD, j = gid % KPAD;
    float v = 0.0f;
    if (j < 3) v = inputs[(size_t)b * TSTEPS * 3 + j];   // x at t=0
    act[gid] = v;
  }
  if (gid < BATCH * ULSTM) cstate[gid] = 0.0f;
  if (gid < BATCH * KGAUSS) kstate[gid] = 0.0f;
}

// -------------------------------------------------------------------------
// z = act @ [Wx;Wh] + b   via v_wmma_f32_16x16x32_bf16
// Grid: 115 blocks x 256 threads = 920 waves; wave (mt, nt) owns a 16x16
// f32 tile of z. A-fragment gathered from fp32 act rows (one aligned
// 16-float run per lane) and converted to bf16; B-fragment is one
// contiguous 32-byte load from wpack.
// -------------------------------------------------------------------------
__global__ __launch_bounds__(256) void gemm_z(
    const __bf16* __restrict__ wpack, const float* __restrict__ act,
    const float* __restrict__ bias, float* __restrict__ z) {
  const int wid  = blockIdx.x * 8 + (threadIdx.x >> 5);   // 0..919 exactly
  const int lane = threadIdx.x & 31;
  const int mt = wid / NT;
  const int nt = wid % NT;
  const int half = lane >> 4;          // 0: lanes 0-15, 1: lanes 16-31
  const int lan  = lane & 15;

  v8f acc = {};
  // A fragment source: row m, k run [base, base+16)
  const float*  arow = act + (size_t)(mt * 16 + lan) * KPAD + half * 16;
  const __bf16* wp   = wpack + (size_t)nt * KC * 512 + lane * 16;

  for (int kc = 0; kc < KC; ++kc) {
    const float4* ap = (const float4*)(arow + kc * 32);
    float4 a0 = ap[0];   // k + 0..3
    float4 a1 = ap[1];   // k + 4..7
    float4 a2 = ap[2];   // k + 8..11
    float4 a3 = ap[3];   // k + 12..15
    v16bf aF;
    aF[0]  = (__bf16)a0.x; aF[1]  = (__bf16)a0.y;
    aF[2]  = (__bf16)a0.z; aF[3]  = (__bf16)a0.w;
    aF[4]  = (__bf16)a1.x; aF[5]  = (__bf16)a1.y;
    aF[6]  = (__bf16)a1.z; aF[7]  = (__bf16)a1.w;
    aF[8]  = (__bf16)a2.x; aF[9]  = (__bf16)a2.y;
    aF[10] = (__bf16)a2.z; aF[11] = (__bf16)a2.w;
    aF[12] = (__bf16)a3.x; aF[13] = (__bf16)a3.y;
    aF[14] = (__bf16)a3.z; aF[15] = (__bf16)a3.w;

    v16bf bF = *(const v16bf*)(wp + (size_t)kc * 512);

    acc = __builtin_amdgcn_wmma_f32_16x16x32_bf16(
        /*neg_a=*/false, aF, /*neg_b=*/false, bF,
        /*c_mod=*/(short)0, acc, /*reuse_a=*/false, /*reuse_b=*/false);
  }

  const int col = nt * 16 + lan;
  if (col < N4) {
    const float bv = bias[col];
    const int rbase = mt * 16 + half * 8;   // C layout: VGPR r -> M = r + 8*half
#pragma unroll
    for (int r = 0; r < 8; ++r)
      z[(size_t)(rbase + r) * N4 + col] = acc[r] + bv;
  }
}

// -------------------------------------------------------------------------
// Pointwise LSTM cell + Gaussian window, one block per batch row.
// -------------------------------------------------------------------------
__global__ __launch_bounds__(256) void cell_window(
    const float* __restrict__ z, const float* __restrict__ Ww,
    const float* __restrict__ bw, const float* __restrict__ sent,
    const float* __restrict__ inputs,
    float* __restrict__ cstate, float* __restrict__ kstate,
    float* __restrict__ act,
    float* __restrict__ hs, float* __restrict__ phis, float* __restrict__ ws,
    int t) {
  const int b   = blockIdx.x;
  const int tid = threadIdx.x;
  __shared__ float hbuf[ULSTM];
  __shared__ float win[32];
  __shared__ float alpha[KGAUSS], beta[KGAUSS], kap[KGAUSS];
  __shared__ float phibuf[SLEN];

  const float* zb   = z + (size_t)b * N4;
  float*       cb   = cstate + (size_t)b * ULSTM;
  float*       actb = act + (size_t)b * KPAD;

  // LSTM pointwise (keras gate order i, f, g, o)
  for (int j = tid; j < ULSTM; j += 256) {
    float iv = zb[j], fv = zb[ULSTM + j], gv = zb[2 * ULSTM + j],
          ov = zb[3 * ULSTM + j];
    float si = 1.0f / (1.0f + __expf(-iv));
    float sf = 1.0f / (1.0f + __expf(-fv));
    float so = 1.0f / (1.0f + __expf(-ov));
    float cn = sf * cb[j] + si * tanhf(gv);
    float hn = so * tanhf(cn);
    cb[j]   = cn;
    hbuf[j] = hn;
    hs[((size_t)b * TSTEPS + t) * ULSTM + j] = hn;
    actb[INDIM + j] = hn;                       // h for next step's GEMM
  }
  __syncthreads();

  // win = h @ Ww + bw   (30 outputs; 8 waves, wave-shuffle reduction)
  const int wv = tid >> 5, lane = tid & 31;
  for (int m = wv; m < 3 * KGAUSS; m += 8) {
    float s = 0.0f;
    for (int j = lane; j < ULSTM; j += 32) s += hbuf[j] * Ww[(size_t)j * 30 + m];
#pragma unroll
    for (int off = 16; off > 0; off >>= 1) s += __shfl_xor(s, off, 32);
    if (lane == 0) win[m] = s + bw[m];
  }
  __syncthreads();

  // alpha, beta, kappa update
  if (tid < KGAUSS) {
    float a  = __expf(win[tid]);
    float be = __expf(win[KGAUSS + tid]);
    float kn = kstate[(size_t)b * KGAUSS + tid] + __expf(win[2 * KGAUSS + tid]);
    kstate[(size_t)b * KGAUSS + tid] = kn;
    alpha[tid] = a; beta[tid] = be; kap[tid] = kn;
  }
  __syncthreads();

  // phi over S=64
  if (tid < SLEN) {
    float u = (float)(tid + 1);
    float p = 0.0f;
#pragma unroll
    for (int kk = 0; kk < KGAUSS; ++kk) {
      float d = kap[kk] - u;
      p += alpha[kk] * __expf(-beta[kk] * d * d);
    }
    phibuf[tid] = p;
    phis[((size_t)b * TSTEPS + t) * SLEN + tid] = p;
  }
  __syncthreads();

  // w_new = phi . sentence  (57 outputs)
  if (tid < NCHARS) {
    const float* sb = sent + (size_t)b * SLEN * NCHARS;
    float s = 0.0f;
    for (int ss = 0; ss < SLEN; ++ss) s += phibuf[ss] * sb[(size_t)ss * NCHARS + tid];
    ws[((size_t)b * TSTEPS + t) * NCHARS + tid] = s;
    actb[3 + tid] = s;                          // w for next step's GEMM
  }

  // stage x_{t+1}
  if (t + 1 < TSTEPS && tid < 3) {
    actb[tid] = inputs[((size_t)b * TSTEPS + (t + 1)) * 3 + tid];
  }
}

// -------------------------------------------------------------------------
extern "C" void kernel_launch(void* const* d_in, const int* in_sizes, int n_in,
                              void* d_out, int out_size, void* d_ws, size_t ws_size,
                              hipStream_t stream) {
  const float* inputs = (const float*)d_in[0];   // [B,T,3]
  const float* sent   = (const float*)d_in[1];   // [B,S,C]
  // d_in[2] = sentence_lengths (unused by the reference math)
  const float* Wx   = (const float*)d_in[3];     // [60, 1828]
  const float* Wh   = (const float*)d_in[4];     // [457, 1828]
  const float* bias = (const float*)d_in[5];     // [1828]
  const float* Ww   = (const float*)d_in[6];     // [457, 30]
  const float* bw   = (const float*)d_in[7];     // [30]

  float* hs   = (float*)d_out;                              // [B,T,457]
  float* phis = hs + (size_t)BATCH * TSTEPS * ULSTM;        // [B,T,64]
  float* wsO  = phis + (size_t)BATCH * TSTEPS * SLEN;       // [B,T,57]

  // workspace carve-up (all offsets 256B-aligned)
  char* base = (char*)d_ws;
  __bf16* wpack  = (__bf16*)(base);                               // 2,001,920 B
  float*  act    = (float*)(base + 2001920);                      //   278,528 B
  float*  zbuf   = (float*)(base + 2001920 + 278528);             //   935,936 B
  float*  cstate = (float*)(base + 2001920 + 278528 + 935936);    //   233,984 B
  float*  kstate = (float*)(base + 2001920 + 278528 + 935936 + 233984); // 5,120 B

  // one-time per launch: weight packing + state init
  {
    int packThreads = NT * KC * 32;                 // 62,560
    int packBlocks  = (packThreads + 255) / 256;
    pack_weights<<<packBlocks, 256, 0, stream>>>(Wx, Wh, wpack);
    int initBlocks = (BATCH * KPAD + 255) / 256;    // covers act, c, kappa
    init_state<<<initBlocks, 256, 0, stream>>>(inputs, act, cstate, kstate);
  }

  for (int t = 0; t < TSTEPS; ++t) {
    gemm_z<<<NT, 256, 0, stream>>>(wpack, act, bias, zbuf);   // 920 WMMA waves
    cell_window<<<BATCH, 256, 0, stream>>>(zbuf, Ww, bw, sent, inputs,
                                           cstate, kstate, act,
                                           hs, phis, wsO, t);
  }
}
